// CrossDiTBlock_29394756174283
// MI455X (gfx1250) — compile-verified, matrix operand order checked by reading, not run on hardware
//
#include <hip/hip_runtime.h>
#include <hip/hip_bf16.h>
#include <math.h>

typedef __attribute__((ext_vector_type(16))) _Float16 v16h;
typedef __attribute__((ext_vector_type(8)))  _Float16 v8h;
typedef __attribute__((ext_vector_type(4)))  _Float16 v4h;
typedef __attribute__((ext_vector_type(8)))  float    v8f;
typedef __attribute__((ext_vector_type(4)))  float    v4f;

#define B_    4
#define N_    1024
#define NC_   77
#define D_    1024
#define H_    16
#define HD_   64
#define DFF_  4096
#define SIXD_ 6144
#define LTOT_ 1101   // N + NC
#define LP_   1120   // padded joint length (multiple of 32)

static __device__ __forceinline__ v16h mk16(v8h lo, v8h hi) {
  v16h r;
#pragma unroll
  for (int i = 0; i < 8; ++i) { r[i] = lo[i]; r[i + 8] = hi[i]; }
  return r;
}

// Async global->LDS copy of 16 bytes (per-lane), tracked by ASYNCcnt.
// LDS address = low 32 bits of the flat pointer (ISA 10.2).
static __device__ __forceinline__ void async_ld_b128(const _Float16* gptr, _Float16* lptr) {
  unsigned lds = (unsigned)(uintptr_t)lptr;
  unsigned long long ga = (unsigned long long)(uintptr_t)gptr;
  asm volatile("global_load_async_to_lds_b128 %0, %1, off" :: "v"(lds), "v"(ga) : "memory");
}
static __device__ __forceinline__ void wait_async0() {
  asm volatile("s_wait_asynccnt 0" ::: "memory");
}

// ---------------------------------------------------------------------------
// adaLN: mod[sel][b][0:6D] = silu(c[b]) @ W + bias   (sel 0: x-branch, 1: cond)
// ---------------------------------------------------------------------------
__global__ __launch_bounds__(256) void ada_kernel(
    const float* __restrict__ c,
    const float* __restrict__ Wa,  const float* __restrict__ ba,
    const float* __restrict__ Wac, const float* __restrict__ bac,
    float* __restrict__ mod)
{
  __shared__ float s[D_];
  int b = blockIdx.y, sel = blockIdx.z, tid = threadIdx.x;
  const float* W  = sel ? Wac : Wa;
  const float* bb = sel ? bac : ba;
#pragma unroll
  for (int i = 0; i < D_ / 256; ++i) {
    int j = i * 256 + tid;
    float v = c[b * D_ + j];
    s[j] = v / (1.0f + __expf(-v));
  }
  __syncthreads();
  int col = blockIdx.x * 256 + tid;
  float acc = 0.f;
  for (int k = 0; k < D_; ++k) acc += s[k] * W[(size_t)k * SIXD_ + col];
  mod[(size_t)sel * (B_ * SIXD_) + (size_t)b * SIXD_ + col] = acc + bb[col];
}

// ---------------------------------------------------------------------------
// LayerNorm + modulate -> f16 row  (one row of D=1024 per 256-thread block)
// ---------------------------------------------------------------------------
__global__ __launch_bounds__(256) void ln_mod_kernel(
    const float* __restrict__ in, const float* __restrict__ modp,
    int sh_off, int sc_off, int rpb,
    _Float16* __restrict__ out)
{
  int row = blockIdx.x, tid = threadIdx.x;
  int batch = row / rpb;
  const float* x = in + (size_t)row * D_;
  float v[4]; float sum = 0.f, sq = 0.f;
#pragma unroll
  for (int i = 0; i < 4; ++i) { v[i] = x[i * 256 + tid]; sum += v[i]; sq += v[i] * v[i]; }
#pragma unroll
  for (int m = 16; m >= 1; m >>= 1) { sum += __shfl_xor(sum, m, 32); sq += __shfl_xor(sq, m, 32); }
  __shared__ float ssum[8], ssq[8];
  int w = tid >> 5, ln = tid & 31;
  if (ln == 0) { ssum[w] = sum; ssq[w] = sq; }
  __syncthreads();
  float ts = 0.f, tq = 0.f;
#pragma unroll
  for (int i = 0; i < 8; ++i) { ts += ssum[i]; tq += ssq[i]; }
  float mu  = ts * (1.0f / D_);
  float var = tq * (1.0f / D_) - mu * mu;
  float rstd = rsqrtf(var + 1e-6f);
  const float* mrow = modp + (size_t)batch * SIXD_;
#pragma unroll
  for (int i = 0; i < 4; ++i) {
    int col = i * 256 + tid;
    float y = (v[i] - mu) * rstd;
    y = y * (1.0f + mrow[sc_off + col]) + mrow[sh_off + col];
    out[(size_t)row * D_ + col] = (_Float16)y;
  }
}

// ---------------------------------------------------------------------------
// Generic WMMA GEMM: C = epilogue( A_f16[M,K] @ W_f32->f16[K,N] + bias )
// 256 threads = 8 waves; macro tile 128x128; wave tile 32x64 (2x4 WMMA tiles)
// A tile staged via GLOBAL_LOAD_ASYNC_TO_LDS_B128 (ASYNCcnt); W tile staged
// as per-thread 4x4 sub-tiles: 4x global b128 loads -> register transpose ->
// 4x ds_store_b64 into [n][k] layout.
// modes: 1 = f16 out; 2 = gelu f16 out; 3 = f32 out = resid + gate*(acc+bias)
// Logical row r -> physical row = (r/rpb)*stride + (r%rpb) for A, C, resid.
// ---------------------------------------------------------------------------
__global__ __launch_bounds__(256) void gemm_wmma_kernel(
    const _Float16* __restrict__ A, const float* __restrict__ W,
    const float* __restrict__ bias,
    const float* __restrict__ gatep, int goff,
    const float* __restrict__ resid,
    float* __restrict__ Cf, _Float16* __restrict__ Ch,
    int M, int Ncols, int K,
    int rpb, int strideA, int strideC, int mode)
{
  __shared__ _Float16 As[128 * 40];   // [row][k], padded stride 40 halfs (80B)
  __shared__ _Float16 Bs[128 * 40];   // [n][k] transposed, padded
  int tid = threadIdx.x;
  int wave = tid >> 5, lane = tid & 31;
  int ln15 = lane & 15, hi8 = (lane >> 4) << 3;
  int wm0 = (wave >> 1) * 32, wn0 = (wave & 1) * 64;
  int m0 = blockIdx.y * 128, n0 = blockIdx.x * 128;

  // ---- loop-invariant staging addresses (strength-reduced) ----
  // A: 2 slots/thread, row = m0 + it*64 + (tid>>2), 8-half chunk at (tid&3)*8
  int ar0 = tid >> 2, ac = (tid & 3) << 3;
  const _Float16* asrc[2];
  _Float16* adst[2];
  bool aval[2];
#pragma unroll
  for (int it = 0; it < 2; ++it) {
    int grow = m0 + it * 64 + ar0;
    aval[it] = (grow < M);
    adst[it] = &As[(it * 64 + ar0) * 40 + ac];
    if (aval[it]) {
      int bb = grow / rpb, rin = grow % rpb;
      asrc[it] = A + ((size_t)bb * strideA + rin) * (size_t)K + ac;
    } else {
      asrc[it] = A;                       // never transferred
      *(v8h*)adst[it] = (v8h){};          // pre-zero guarded rows once
    }
  }
  // W: per-thread 4x4 sub-tile, k rows = (tid>>5)*4..+3, n cols = (tid&31)*4..+3
  int wrow = (tid >> 5) << 2;       // 0..28
  int wcol = (tid & 31) << 2;       // 0..124
  const float* wbase = W + (size_t)wrow * Ncols + n0 + wcol;

  v8f acc[2][4];
#pragma unroll
  for (int i = 0; i < 2; ++i)
#pragma unroll
    for (int j = 0; j < 4; ++j) acc[i][j] = (v8f){};

  for (int k0 = 0; k0 < K; k0 += 32) {
    // async-stage A tile 128x32 f16 (EXEC-masked for guarded rows)
#pragma unroll
    for (int it = 0; it < 2; ++it)
      if (aval[it]) async_ld_b128(asrc[it], adst[it]);
    // stage W tile 32x128: 4x coalesced b128 f32 loads -> 4x4 transpose -> b64 LDS
    {
      v4f lk0 = *(const v4f*)(wbase);
      v4f lk1 = *(const v4f*)(wbase + (size_t)Ncols);
      v4f lk2 = *(const v4f*)(wbase + (size_t)2 * Ncols);
      v4f lk3 = *(const v4f*)(wbase + (size_t)3 * Ncols);
#pragma unroll
      for (int j = 0; j < 4; ++j) {
        v4h p;
        p[0] = (_Float16)lk0[j]; p[1] = (_Float16)lk1[j];
        p[2] = (_Float16)lk2[j]; p[3] = (_Float16)lk3[j];
        *(v4h*)&Bs[(wcol + j) * 40 + wrow] = p;
      }
    }
    if (k0 + 32 < K) __builtin_prefetch(wbase + (size_t)32 * Ncols, 0, 0);
    wait_async0();
    __syncthreads();

    v16h afr[2], bfr[4];
#pragma unroll
    for (int mt = 0; mt < 2; ++mt) {
      const _Float16* p = &As[(wm0 + mt * 16 + ln15) * 40];
      afr[mt] = mk16(*(const v8h*)(p + hi8), *(const v8h*)(p + 16 + hi8));
    }
#pragma unroll
    for (int nt = 0; nt < 4; ++nt) {
      const _Float16* p = &Bs[(wn0 + nt * 16 + ln15) * 40];
      bfr[nt] = mk16(*(const v8h*)(p + hi8), *(const v8h*)(p + 16 + hi8));
    }
#pragma unroll
    for (int mt = 0; mt < 2; ++mt)
#pragma unroll
      for (int nt = 0; nt < 4; ++nt)
        acc[mt][nt] = __builtin_amdgcn_wmma_f32_16x16x32_f16(
            false, afr[mt], false, bfr[nt], (short)0, acc[mt][nt], false, false);
    __syncthreads();

    asrc[0] += 32; asrc[1] += 32;
    wbase += (size_t)32 * Ncols;
  }

  // epilogue: C layout lane holds col = ln15, rows r + 8*(lane>>4)
#pragma unroll
  for (int mt = 0; mt < 2; ++mt) {
#pragma unroll
    for (int nt = 0; nt < 4; ++nt) {
#pragma unroll
      for (int r = 0; r < 8; ++r) {
        int gm = m0 + wm0 + mt * 16 + r + hi8;
        int gn = n0 + wn0 + nt * 16 + ln15;
        if (gm < M) {
          int bb = gm / rpb, rin = gm % rpb;
          size_t crow = (size_t)bb * strideC + rin;
          float vacc = acc[mt][nt][r] + bias[gn];
          if (mode == 1) {
            Ch[crow * Ncols + gn] = (_Float16)vacc;
          } else if (mode == 2) {
            float t = 0.7978845608f * (vacc + 0.044715f * vacc * vacc * vacc);
            float g = 0.5f * vacc * (1.0f + tanhf(t));
            Ch[crow * Ncols + gn] = (_Float16)g;
          } else {
            float g = gatep[(size_t)bb * SIXD_ + goff + gn];
            Cf[crow * Ncols + gn] = resid[crow * Ncols + gn] + g * vacc;
          }
        }
      }
    }
  }
}

// ---------------------------------------------------------------------------
// Repack qkv f16 -> Q (scaled), K row-major [b,h,tok,d]; V transposed [b,h,d,tok]
// Writes zeros for padded tokens (LTOT_..LP_) every call.
// ---------------------------------------------------------------------------
__global__ __launch_bounds__(256) void repack_qkv_kernel(
    const _Float16* __restrict__ qkvx, const _Float16* __restrict__ qkvc,
    _Float16* __restrict__ Qh, _Float16* __restrict__ Kh, _Float16* __restrict__ Vt)
{
  int idx = blockIdx.x * 256 + threadIdx.x;
  const int tot = B_ * LP_ * D_;
  if (idx >= tot) return;
  int b   = idx / (LP_ * D_);
  int rem = idx % (LP_ * D_);
  int tok = rem / D_;
  int col = rem % D_;
  int h = col >> 6, d = col & 63;
  float q = 0.f, k = 0.f, vv = 0.f;
  if (tok < N_) {
    const _Float16* s = qkvx + ((size_t)(b * N_ + tok)) * (3 * D_);
    q = (float)s[col]; k = (float)s[D_ + col]; vv = (float)s[2 * D_ + col];
  } else if (tok < LTOT_) {
    const _Float16* s = qkvc + ((size_t)(b * NC_ + (tok - N_))) * (3 * D_);
    q = (float)s[col]; k = (float)s[D_ + col]; vv = (float)s[2 * D_ + col];
  }
  size_t rq = ((size_t)(b * H_ + h) * LP_ + tok) * HD_ + d;
  Qh[rq] = (_Float16)(q * 0.125f);   // HD^-0.5
  Kh[rq] = (_Float16)k;
  Vt[((size_t)(b * H_ + h) * HD_ + d) * LP_ + tok] = (_Float16)vv;
}

// ---------------------------------------------------------------------------
// Flash attention: 1 wave per (b, h, 16-row q-tile) over joint length LP_.
// S = Q K^T via WMMA (B-frag of K^T == rows of K); online softmax; P via LDS
// re-layout (C->A); O += P V using V^T fragments. Output f16 [b, tok, h*64+d].
// ---------------------------------------------------------------------------
__global__ __launch_bounds__(32) void attn_kernel(
    const _Float16* __restrict__ Qh, const _Float16* __restrict__ Kh,
    const _Float16* __restrict__ Vt, _Float16* __restrict__ Obuf)
{
  __shared__ _Float16 plds[16 * 40];
  int lane = threadIdx.x & 31;
  int ln15 = lane & 15, hi8 = (lane >> 4) << 3;
  int qt = blockIdx.x, h = blockIdx.y, b = blockIdx.z;
  int q0 = qt * 16;
  const size_t bh = (size_t)(b * H_ + h);

  const _Float16* qrow = Qh + (bh * LP_ + (q0 + ln15)) * HD_;
  v16h aq[2];
#pragma unroll
  for (int c = 0; c < 2; ++c)
    aq[c] = mk16(*(const v8h*)(qrow + c * 32 + hi8),
                 *(const v8h*)(qrow + c * 32 + 16 + hi8));

  v8f o[4];
#pragma unroll
  for (int i = 0; i < 4; ++i) o[i] = (v8f){};
  float mrow[8], lrow[8];
#pragma unroll
  for (int r = 0; r < 8; ++r) { mrow[r] = -3.0e38f; lrow[r] = 0.f; }

  for (int kb = 0; kb < LP_; kb += 32) {
    v8f s[2];
#pragma unroll
    for (int t = 0; t < 2; ++t) {
      const _Float16* krow = Kh + (bh * LP_ + (kb + t * 16 + ln15)) * HD_;
      v8f z = (v8f){};
#pragma unroll
      for (int c = 0; c < 2; ++c) {
        v16h bk = mk16(*(const v8h*)(krow + c * 32 + hi8),
                       *(const v8h*)(krow + c * 32 + 16 + hi8));
        z = __builtin_amdgcn_wmma_f32_16x16x32_f16(false, aq[c], false, bk,
                                                   (short)0, z, false, false);
      }
      if (kb + t * 16 + ln15 >= LTOT_) {   // mask padded keys (lane == key col)
#pragma unroll
        for (int r = 0; r < 8; ++r) z[r] = -1.0e30f;
      }
      s[t] = z;
    }
    // online softmax per row (row = r + hi8; stats reduced over 16-lane group)
    float mnew[8], alpha[8];
#pragma unroll
    for (int r = 0; r < 8; ++r) {
      float tm = fmaxf(s[0][r], s[1][r]);
#pragma unroll
      for (int m = 8; m >= 1; m >>= 1) tm = fmaxf(tm, __shfl_xor(tm, m, 32));
      mnew[r] = fmaxf(mrow[r], tm);
      alpha[r] = __expf(mrow[r] - mnew[r]);
    }
#pragma unroll
    for (int r = 0; r < 8; ++r) {
      float p0 = __expf(s[0][r] - mnew[r]);
      float p1 = __expf(s[1][r] - mnew[r]);
      float rs = p0 + p1;
#pragma unroll
      for (int m = 8; m >= 1; m >>= 1) rs += __shfl_xor(rs, m, 32);
      lrow[r] = lrow[r] * alpha[r] + rs;
      mrow[r] = mnew[r];
      plds[(r + hi8) * 40 + ln15]      = (_Float16)p0;
      plds[(r + hi8) * 40 + 16 + ln15] = (_Float16)p1;
#pragma unroll
      for (int nt = 0; nt < 4; ++nt) o[nt][r] *= alpha[r];
    }
    asm volatile("s_wait_dscnt 0" ::: "memory");  // LDS in-order within wave
    const _Float16* prow = &plds[ln15 * 40];
    v16h ap = mk16(*(const v8h*)(prow + hi8), *(const v8h*)(prow + 16 + hi8));
#pragma unroll
    for (int nt = 0; nt < 4; ++nt) {
      const _Float16* vrow = Vt + (bh * HD_ + (nt * 16 + ln15)) * LP_ + kb;
      v16h bv = mk16(*(const v8h*)(vrow + hi8), *(const v8h*)(vrow + 16 + hi8));
      o[nt] = __builtin_amdgcn_wmma_f32_16x16x32_f16(false, ap, false, bv,
                                                     (short)0, o[nt], false, false);
    }
  }
  // normalize and store
#pragma unroll
  for (int r = 0; r < 8; ++r) {
    int tok = q0 + r + hi8;
    if (tok < LTOT_) {
      float inv = 1.0f / lrow[r];
#pragma unroll
      for (int nt = 0; nt < 4; ++nt) {
        int col = h * HD_ + nt * 16 + ln15;
        Obuf[((size_t)b * LP_ + tok) * D_ + col] = (_Float16)(o[nt][r] * inv);
      }
    }
  }
}

// ---------------------------------------------------------------------------
extern "C" void kernel_launch(void* const* d_in, const int* in_sizes, int n_in,
                              void* d_out, int out_size, void* d_ws, size_t ws_size,
                              hipStream_t stream)
{
  const float* x       = (const float*)d_in[0];
  const float* cond    = (const float*)d_in[1];
  const float* c       = (const float*)d_in[2];
  const float* W_ada   = (const float*)d_in[3];
  const float* b_ada   = (const float*)d_in[4];
  const float* W_ada_c = (const float*)d_in[5];
  const float* b_ada_c = (const float*)d_in[6];
  const float* W_qkv   = (const float*)d_in[7];
  const float* b_qkv   = (const float*)d_in[8];
  const float* W_proj  = (const float*)d_in[9];
  const float* b_proj  = (const float*)d_in[10];
  const float* W_qkv_c = (const float*)d_in[11];
  const float* b_qkv_c = (const float*)d_in[12];
  const float* W_proj_c= (const float*)d_in[13];
  const float* b_proj_c= (const float*)d_in[14];
  const float* W_fc1   = (const float*)d_in[15];
  const float* b_fc1   = (const float*)d_in[16];
  const float* W_fc2   = (const float*)d_in[17];
  const float* b_fc2   = (const float*)d_in[18];
  const float* W_fc1_c = (const float*)d_in[19];
  const float* b_fc1_c = (const float*)d_in[20];
  const float* W_fc2_c = (const float*)d_in[21];
  const float* b_fc2_c = (const float*)d_in[22];

  char* ws = (char*)d_ws;
  size_t off = 0;
  auto alloc = [&](size_t bytes) { size_t o = off; off += (bytes + 255) & ~(size_t)255; return o; };

  float*     mod  = (float*)    (ws + alloc((size_t)2 * B_ * SIXD_ * 4));
  _Float16*  xA   = (_Float16*) (ws + alloc((size_t)B_ * N_  * D_   * 2));
  _Float16*  cA   = (_Float16*) (ws + alloc((size_t)B_ * NC_ * D_   * 2));
  _Float16*  h1   = (_Float16*) (ws + alloc((size_t)B_ * N_  * DFF_ * 2)); // qkvx aliases
  _Float16*  h1c  = (_Float16*) (ws + alloc((size_t)B_ * NC_ * DFF_ * 2)); // qkvc aliases
  _Float16*  Qh   = (_Float16*) (ws + alloc((size_t)B_ * H_ * LP_ * HD_ * 2));
  _Float16*  Kh   = (_Float16*) (ws + alloc((size_t)B_ * H_ * LP_ * HD_ * 2));
  _Float16*  Vt   = (_Float16*) (ws + alloc((size_t)B_ * H_ * HD_ * LP_ * 2));
  _Float16*  Obuf = (_Float16*) (ws + alloc((size_t)B_ * LP_ * D_ * 2));
  float*     xmid = (float*)    (ws + alloc((size_t)B_ * N_  * D_ * 4));
  float*     cmid = (float*)    (ws + alloc((size_t)B_ * NC_ * D_ * 4));
  _Float16*  qkvx = h1;   // [B*N, 3D]  (dead before fc1 writes h1)
  _Float16*  qkvc = h1c;  // [B*NC, 3D]

  const float* mod_x = mod;
  const float* mod_c = mod + (size_t)B_ * SIXD_;
  const int Mx = B_ * N_;   // 4096
  const int Mc = B_ * NC_;  // 308

  // 1) adaLN modulation vectors
  ada_kernel<<<dim3(SIXD_ / 256, B_, 2), 256, 0, stream>>>(
      c, W_ada, b_ada, W_ada_c, b_ada_c, mod);

  // 2) LN + modulate (MSA params: shift @0, scale @D)
  ln_mod_kernel<<<Mx, 256, 0, stream>>>(x,    mod_x, 0, D_, N_,  xA);
  ln_mod_kernel<<<Mc, 256, 0, stream>>>(cond, mod_c, 0, D_, NC_, cA);

  // 3) QKV GEMMs (f16 out)
  gemm_wmma_kernel<<<dim3(3 * D_ / 128, Mx / 128), 256, 0, stream>>>(
      xA, W_qkv, b_qkv, nullptr, 0, nullptr, nullptr, qkvx,
      Mx, 3 * D_, D_, N_, N_, N_, 1);
  gemm_wmma_kernel<<<dim3(3 * D_ / 128, (Mc + 127) / 128), 256, 0, stream>>>(
      cA, W_qkv_c, b_qkv_c, nullptr, 0, nullptr, nullptr, qkvc,
      Mc, 3 * D_, D_, NC_, NC_, NC_, 1);

  // 4) repack Q/K/V (covers all LP_ tokens incl. zero pads)
  repack_qkv_kernel<<<(B_ * LP_ * D_) / 256, 256, 0, stream>>>(qkvx, qkvc, Qh, Kh, Vt);

  // 5) joint flash attention
  attn_kernel<<<dim3(LP_ / 16, H_, B_), 32, 0, stream>>>(Qh, Kh, Vt, Obuf);

  // 6) output projections + gate*.. + residual
  gemm_wmma_kernel<<<dim3(D_ / 128, Mx / 128), 256, 0, stream>>>(
      Obuf, W_proj, b_proj, mod_x, 2 * D_, x, xmid, nullptr,
      Mx, D_, D_, N_, LP_, N_, 3);
  gemm_wmma_kernel<<<dim3(D_ / 128, (Mc + 127) / 128), 256, 0, stream>>>(
      Obuf + (size_t)N_ * D_, W_proj_c, b_proj_c, mod_c, 2 * D_, cond, cmid, nullptr,
      Mc, D_, D_, NC_, LP_, NC_, 3);

  // 7) second LN + modulate (MLP params: shift @3D, scale @4D)
  ln_mod_kernel<<<Mx, 256, 0, stream>>>(xmid, mod_x, 3 * D_, 4 * D_, N_,  xA);
  ln_mod_kernel<<<Mc, 256, 0, stream>>>(cmid, mod_c, 3 * D_, 4 * D_, NC_, cA);

  // 8) fc1 + GELU (f16 out)
  gemm_wmma_kernel<<<dim3(DFF_ / 128, Mx / 128), 256, 0, stream>>>(
      xA, W_fc1, b_fc1, nullptr, 0, nullptr, nullptr, h1,
      Mx, DFF_, D_, N_, N_, N_, 2);
  gemm_wmma_kernel<<<dim3(DFF_ / 128, (Mc + 127) / 128), 256, 0, stream>>>(
      cA, W_fc1_c, b_fc1_c, nullptr, 0, nullptr, nullptr, h1c,
      Mc, DFF_, D_, NC_, NC_, NC_, 2);

  // 9) fc2 + gate + residual -> d_out (x then cond, concatenated)
  float* out_x = (float*)d_out;
  float* out_c = (float*)d_out + (size_t)B_ * N_ * D_;
  gemm_wmma_kernel<<<dim3(D_ / 128, Mx / 128), 256, 0, stream>>>(
      h1, W_fc2, b_fc2, mod_x, 5 * D_, xmid, out_x, nullptr,
      Mx, D_, DFF_, N_, N_, N_, 3);
  gemm_wmma_kernel<<<dim3(D_ / 128, (Mc + 127) / 128), 256, 0, stream>>>(
      h1c, W_fc2_c, b_fc2_c, mod_c, 5 * D_, cmid, out_c, nullptr,
      Mc, D_, DFF_, NC_, NC_, NC_, 3);
}
